// Attention_48103633715423
// MI455X (gfx1250) — matrix operand order repaired and checked by reading, not compiled, not run on hardware
//
#include <hip/hip_runtime.h>
#include <hip/hip_bf16.h>
#include <stdint.h>

typedef __attribute__((ext_vector_type(16))) __bf16 v16bf;
typedef __attribute__((ext_vector_type(8)))  float  v8f;
typedef __attribute__((ext_vector_type(8)))  unsigned int v8u;
typedef __attribute__((ext_vector_type(4)))  unsigned int v4u;
typedef __attribute__((ext_vector_type(8)))  int v8i;
typedef __attribute__((ext_vector_type(4)))  int v4i;

#define B_SZ 128
#define N_SZ 4096
#define H_SZ 128
#define P_SZ 128
#define NEGV -10000.0f
#define TILES_PER_BLK 4

// ---------------- Kernel 1: weight-norm scales ----------------
__global__ __launch_bounds__(256) void k_norms(
    const float* __restrict__ Wv, const float* __restrict__ gv,
    const float* __restrict__ Wq, const float* __restrict__ gq,
    const float* __restrict__ Wa, const float* __restrict__ ga,
    float* __restrict__ scal) {
  __shared__ float sh0[256], sh1[256], sh2[256];
  const int t = threadIdx.x;
  float s0 = 0.f, s1 = 0.f, s2 = 0.f;
  for (int i = t; i < P_SZ * H_SZ; i += 256) {
    float a = Wv[i]; s0 += a * a;
    float b = Wq[i]; s1 += b * b;
  }
  for (int i = t; i < P_SZ; i += 256) { float c = Wa[i]; s2 += c * c; }
  sh0[t] = s0; sh1[t] = s1; sh2[t] = s2;
  __syncthreads();
  for (int off = 128; off > 0; off >>= 1) {
    if (t < off) { sh0[t] += sh0[t + off]; sh1[t] += sh1[t + off]; sh2[t] += sh2[t + off]; }
    __syncthreads();
  }
  if (t == 0) {
    scal[0] = gv[0] / sqrtf(sh0[0]);
    scal[1] = gq[0] / sqrtf(sh1[0]);
    scal[2] = ga[0] / sqrtf(sh2[0]);
  }
}

// ---------------- Kernel 2: qproj (+bq+bv), wa_s, bf16 Wv ----------------
__global__ __launch_bounds__(128) void k_prep(
    const float* __restrict__ q,  const float* __restrict__ Wq,
    const float* __restrict__ bq, const float* __restrict__ bv,
    const float* __restrict__ Wa, const float* __restrict__ Wv,
    const float* __restrict__ scal,
    float* __restrict__ qproj, float* __restrict__ wa_s,
    __bf16* __restrict__ Wvb) {
  const int bb = blockIdx.x, t = threadIdx.x;
  if (bb < B_SZ) {
    const float* qr = q + bb * H_SZ;
    const float* wr = Wq + t * H_SZ;
    float s = 0.f;
    #pragma unroll 8
    for (int h = 0; h < H_SZ; ++h) s += qr[h] * wr[h];
    qproj[bb * P_SZ + t] = scal[1] * s + bq[t] + bv[t];
  } else if (bb == B_SZ) {
    wa_s[t] = scal[2] * Wa[t];
  } else {
    const int row = bb - (B_SZ + 1);
    Wvb[row * H_SZ + t] = (__bf16)(scal[0] * Wv[row * H_SZ + t]);
  }
}

// ---------------- TDM: DMA one 128x128 f32 tile global -> LDS ----------------
// D# per CDNA5 ISA ch.8: group0 = {flags, lds_addr, global_addr, type=2},
// group1 = {data_size=4B, tensor_dim0/1 = 128, tile_dim0/1 = 128,
//           tensor_dim0_stride = 128}, groups 2/3 unused (2D tile).
__device__ __forceinline__ void tdm_load_tile(const float* gptr, float* ldsptr) {
  const unsigned long long ga = (unsigned long long)(uintptr_t)gptr;
  const unsigned lds = (unsigned)(uintptr_t)ldsptr;   // LDS byte offset
  v4u g0 = { 1u,                                      // count=1 (valid), no gather
             lds,
             (unsigned)(ga & 0xffffffffull),
             (unsigned)((ga >> 32) & 0x1ffffffull) | (2u << 30) };  // type=2
  v8i g1 = { (int)0x00020000,      // data_size = 2 (4 bytes)
             (int)(128u << 16),    // tensor_dim0[15:0]  = 128 (bits 79:48)
             (int)(128u << 16),    // tensor_dim1[15:0]  = 128 (bits 111:80)
             (int)(128u << 16),    // tile_dim0          = 128 (bits 127:112)
             128,                  // tile_dim1          = 128 (bits 143:128)
             128,                  // tensor_dim0_stride = 128 (bits 207:160)
             0, 0 };
  v4i g2 = { 0, 0, 0, 0 };
  v4i g3 = { 0, 0, 0, 0 };
#if __clang_major__ >= 23
  v8i gx = { 0, 0, 0, 0, 0, 0, 0, 0 };
  __builtin_amdgcn_tensor_load_to_lds(g0, g1, g2, g3, gx, 0);
#else
  __builtin_amdgcn_tensor_load_to_lds(g0, g1, g2, g3, 0);
#endif
}

// ---------------- Kernel 3: fused WMMA GEMM -> logits ----------------
// grid = (N/(128*TILES_PER_BLK), B), block = 256 (8 waves).
// Double-buffered TDM: while a 64KB v-tile is consumed by WMMA, the DMA
// engine streams the next tile into the other LDS buffer.
__global__ __launch_bounds__(256) void k_main(
    const float* __restrict__ v,     const float* __restrict__ qproj,
    const float* __restrict__ wa_s,  const __bf16* __restrict__ Wvb,
    const float* __restrict__ mask,  const float* __restrict__ ba,
    float* __restrict__ logits) {
  __shared__ float tile[2][128 * 128];         // 2 x 64 KB double buffer
  const int t  = threadIdx.x;
  const int b  = blockIdx.y;
  const int nt0 = blockIdx.x * TILES_PER_BLK;  // first n-tile index
  const float* vb = v + (size_t)b * N_SZ * H_SZ;

  const int wave = t >> 5, lane = t & 31;
  const int m      = lane & 15;
  const int kbaseA = (lane < 16) ? 0 : 16;   // contiguous k layout (matches B)
  const int kbaseB = (lane < 16) ? 0 : 16;   // contiguous k layout (matches A)
  const int p_col  = lane & 15;
  const float ba0  = ba[0];

  // Prime the pipeline: wave 0 issues the DMA for tile 0.
  if (t < 32) tdm_load_tile(vb + (size_t)(nt0 * 128) * H_SZ, tile[0]);

  for (int it = 0; it < TILES_PER_BLK; ++it) {
    const int buf = it & 1;
    const int n0  = (nt0 + it) * 128;

    // Publish tile `it` (wave 0 drains TENSORcnt), then one barrier both
    // releases the readers and retires the previous consumers of the other
    // buffer, so wave 0 can immediately issue the next DMA.
    if (t < 32) __builtin_amdgcn_s_wait_tensorcnt(0);
    __syncthreads();
    if ((t < 32) && (it + 1 < TILES_PER_BLK))
      tdm_load_tile(vb + (size_t)((nt0 + it + 1) * 128) * H_SZ,
                    tile[(it + 1) & 1]);

    // Scheduling barrier: the opaque asm may "write memory", so the Wvb
    // B-fragment loads cannot be hoisted out of / CSE'd across the tile loop
    // (which would pin 256 VGPRs of B data and halve occupancy), while the
    // pointer keeps its global address space (global_load, not flat_load).
    asm volatile("" ::: "memory");

    // A fragments: 16x32 bf16 per k-chunk, 4 chunks cover K=H=128.
    const float* arow = tile[buf] + (wave * 16 + m) * H_SZ;
    v16bf a[4];
    #pragma unroll
    for (int kc = 0; kc < 4; ++kc) {
      const int k0 = kc * 32;
      #pragma unroll
      for (int e = 0; e < 16; ++e) {
        const int k = k0 + kbaseA + e;
        a[kc][e] = (__bf16)arow[k];
      }
    }

    // 8 independent accumulator chains (one per p-tile) so the scheduler can
    // batch the B-fragment loads ahead of the WMMAs instead of stalling each.
    v8f acc[8];
    #pragma unroll
    for (int pt = 0; pt < 8; ++pt)
      acc[pt] = (v8f){0.f, 0.f, 0.f, 0.f, 0.f, 0.f, 0.f, 0.f};
    #pragma unroll
    for (int kc = 0; kc < 4; ++kc) {
      #pragma unroll
      for (int pt = 0; pt < 8; ++pt) {
        const int p = pt * 16 + p_col;
        // B = Wv_n^T (KxN): element e -> K = kc*32 + kbaseB + e, 32B contig.
        v8u raw = *(const v8u*)(Wvb + (size_t)p * H_SZ + kc * 32 + kbaseB);
        v16bf bfrag = __builtin_bit_cast(v16bf, raw);
        acc[pt] = __builtin_amdgcn_wmma_f32_16x16x32_bf16(
            false, a[kc], false, bfrag, (short)0, acc[pt], false, false);
      }
    }

    // Fused epilogue: h = relu(vproj + qproj), partial dot with wa_s.
    // C layout: lane -> col p, VGPR r -> row M=r (lanes 0-15) or r+8 (16-31).
    float lsum[8] = {0.f, 0.f, 0.f, 0.f, 0.f, 0.f, 0.f, 0.f};
    #pragma unroll
    for (int pt = 0; pt < 8; ++pt) {
      const int p = pt * 16 + p_col;
      const float qv = qproj[b * P_SZ + p];
      const float wa = wa_s[p];
      #pragma unroll
      for (int r = 0; r < 8; ++r) {
        float hv = acc[pt][r] + qv;
        hv = fmaxf(hv, 0.f);
        lsum[r] += hv * wa;
      }
    }

    // Sum over the 16 columns held in each lane-half.
    #pragma unroll
    for (int r = 0; r < 8; ++r) {
      float s = lsum[r];
      s += __shfl_xor(s, 1);
      s += __shfl_xor(s, 2);
      s += __shfl_xor(s, 4);
      s += __shfl_xor(s, 8);
      lsum[r] = s;
    }

    if ((lane & 15) == 0) {
      const int moff = (lane < 16) ? 0 : 8;
      #pragma unroll
      for (int r = 0; r < 8; ++r) {
        const int n = n0 + wave * 16 + r + moff;
        const float lg = lsum[r] + ba0 + (1.0f - mask[b * N_SZ + n]) * NEGV;
        logits[(size_t)b * N_SZ + n] = lg;
      }
    }
    __syncthreads();   // consumers of tile[buf] done before it is re-filled
  }
}

// ---------------- Kernel 4: softmax + weighted sum ----------------
// One block per batch. logits live in the w_att output region; we read them
// fully (passes 1-2) before overwriting with normalized weights (pass 3).
__global__ __launch_bounds__(256) void k_soft(
    const float* __restrict__ v, float* __restrict__ logits_w,
    float* __restrict__ att) {
  __shared__ float wsh[N_SZ];
  __shared__ float red[256];
  const int t = threadIdx.x;
  const int b = blockIdx.x;
  float* lg = logits_w + (size_t)b * N_SZ;

  float mx = -3.4e38f;
  for (int n = t; n < N_SZ; n += 256) mx = fmaxf(mx, lg[n]);
  red[t] = mx; __syncthreads();
  for (int off = 128; off > 0; off >>= 1) {
    if (t < off) red[t] = fmaxf(red[t], red[t + off]);
    __syncthreads();
  }
  mx = red[0]; __syncthreads();

  float s = 0.f;
  for (int n = t; n < N_SZ; n += 256) {
    const float e = expf(lg[n] - mx);
    wsh[n] = e; s += e;
  }
  red[t] = s; __syncthreads();
  for (int off = 128; off > 0; off >>= 1) {
    if (t < off) red[t] += red[t + off];
    __syncthreads();
  }
  const float inv = 1.0f / red[0];
  __syncthreads();

  for (int n = t; n < N_SZ; n += 256) {
    const float w = wsh[n] * inv;
    wsh[n] = w;
    lg[n] = w;                 // final w_att output
  }
  __syncthreads();

  // att[b,h] = sum_n w[n] * v[b,n,h]; coalesced over h.
  const int h = t & 127, part = t >> 7;
  const float* vb = v + (size_t)b * N_SZ * H_SZ;
  float acc = 0.f;
  for (int n = part; n < N_SZ; n += 2)
    acc += wsh[n] * vb[(size_t)n * H_SZ + h];
  red[t] = acc; __syncthreads();
  if (part == 0) att[b * H_SZ + h] = red[t] + red[t + 128];
}

// ---------------- Launcher ----------------
extern "C" void kernel_launch(void* const* d_in, const int* in_sizes, int n_in,
                              void* d_out, int out_size, void* d_ws, size_t ws_size,
                              hipStream_t stream) {
  const float* v    = (const float*)d_in[0];
  const float* q    = (const float*)d_in[1];
  const float* mask = (const float*)d_in[2];
  const float* Wv   = (const float*)d_in[3];
  const float* bv   = (const float*)d_in[4];
  const float* gv   = (const float*)d_in[5];
  const float* Wq   = (const float*)d_in[6];
  const float* bq   = (const float*)d_in[7];
  const float* gq   = (const float*)d_in[8];
  const float* Wa   = (const float*)d_in[9];
  const float* ba   = (const float*)d_in[10];
  const float* ga   = (const float*)d_in[11];

  float* att   = (float*)d_out;                 // (B,H) = 16384 floats
  float* w_out = (float*)d_out + B_SZ * H_SZ;   // (B,N,1) = 524288 floats

  char* wsb = (char*)d_ws;
  float*  scal  = (float*)wsb;                                   // 3 floats (pad 64B)
  float*  qproj = (float*)(wsb + 64);                            // B*P floats
  float*  wa_s  = (float*)(wsb + 64 + B_SZ * P_SZ * 4);          // P floats
  __bf16* Wvb   = (__bf16*)(wsb + 64 + B_SZ * P_SZ * 4 + 512);   // P*H bf16

  k_norms<<<1, 256, 0, stream>>>(Wv, gv, Wq, gq, Wa, ga, scal);
  k_prep<<<B_SZ + 1 + P_SZ, 128, 0, stream>>>(q, Wq, bq, bv, Wa, Wv, scal,
                                              qproj, wa_s, Wvb);
  dim3 g(N_SZ / (128 * TILES_PER_BLK), B_SZ);
  k_main<<<g, 256, 0, stream>>>(v, qproj, wa_s, Wvb, mask, ba, w_out);
  k_soft<<<B_SZ, 256, 0, stream>>>(v, w_out, att);
}
